// SlotAttention_47459388621485
// MI455X (gfx1250) — compile-verified, matrix operand order checked by reading, not run on hardware
//
#include <hip/hip_runtime.h>
#include <math.h>

typedef __attribute__((ext_vector_type(16))) _Float16 v16h;
typedef __attribute__((ext_vector_type(8)))  _Float16 v8h;
typedef __attribute__((ext_vector_type(8)))  float    v8f;

constexpr int Bc = 32, Nc = 4096, Fc = 256, Sc = 256, Mc = 1024, Kc = 11;
constexpr int Rc = Bc * Kc;           // 352 slot rows total
constexpr float SLOT_EPS = 1e-6f;
constexpr float LN_EPS   = 1e-5f;

constexpr int NT = 4;                 // N-tiles (16 cols each) per wave -> 64 cols

__device__ __forceinline__ void load_frags(const _Float16* aPtr,
                                           const _Float16* const* bPtr,
                                           int k0, v16h& a, v16h* b) {
  const v8h* ap = (const v8h*)(aPtr + k0);
  v8h a0 = ap[0], a1 = ap[2];
#pragma unroll
  for (int j = 0; j < 8; ++j) { a[j] = a0[j]; a[8 + j] = a1[j]; }
#pragma unroll
  for (int u = 0; u < NT; ++u) {
    const v8h* bp = (const v8h*)(bPtr[u] + k0);
    v8h b0 = bp[0], b1 = bp[2];
#pragma unroll
    for (int j = 0; j < 8; ++j) { b[u][j] = b0[j]; b[u][8 + j] = b1[j]; }
  }
}

// ---------------------------------------------------------------------------
// Batched WMMA GEMM: C[b] = alpha * A[b] x B[b]^T (+bias)(+relu)(+residual)
//   A : f16 [M x Kd] row-major (lda)       (rows clamped, no hot-loop guards)
//   B : f16 [N x Kd] row-major (ldb)       (C[i,j] = sum_d A[i,d]*B[j,d])
//   Out: f32 -> C (row-major), or f16 -> Ch (row-major, or transposed
//        [N x M] with leading dim ldcT when storeT=1).
// One wave per (16-row M tile) x (64-col N group). Software-pipelined:
// ping-pong fragment buffers, K unrolled by 64, so each 4-WMMA burst overlaps
// with the in-flight load clause of the next K step.
// Requirements (host guarantees): N % 64 == 0, Kd % 64 == 0, M >= 1.
// ---------------------------------------------------------------------------
__global__ void wmma_gemm_f16_kernel(
    const _Float16* __restrict__ A, int lda, long long strideA,
    const _Float16* __restrict__ Bm, int ldb, long long strideB,
    float* __restrict__ C, _Float16* __restrict__ Ch, int ldc, long long strideC,
    int M, int Kd, int tilesM, int groupsN,
    float alpha, const float* __restrict__ bias, int doRelu,
    const float* __restrict__ residual,
    int storeT, int ldcT)
{
  int tilesPer = tilesM * groupsN;
  int gid   = blockIdx.x;
  int batch = gid / tilesPer;
  int t     = gid - batch * tilesPer;
  int tm    = t / groupsN;
  int tg    = t - tm * groupsN;
  int n0    = tg * (NT * 16);

  const _Float16* Ab = A  + (size_t)batch * strideA;
  const _Float16* Bb = Bm + (size_t)batch * strideB;

  int lane = threadIdx.x & 31;
  int hi   = lane >> 4;                       // lane half
  int l16  = lane & 15;
  int kb   = hi * 8;                          // K sub-block base in 32-wide step
  int arow = tm * 16 + l16;
  int arowL = arow < M ? arow : (M - 1);      // clamp: OOB rows recompute row M-1,
                                              // epilogue m<M guard discards them
  const _Float16* aPtr = Ab + (size_t)arowL * lda + kb;
  const _Float16* bPtr[NT];
#pragma unroll
  for (int u = 0; u < NT; ++u)
    bPtr[u] = Bb + (size_t)(n0 + u * 16 + l16) * ldb + kb;

  v8f acc[NT] = {};
  v16h a0f, a1f;
  v16h b0f[NT], b1f[NT];

  // prologue: fill buffer 0, prefetch one step ahead (speculative)
  load_frags(aPtr, bPtr, 0, a0f, b0f);
  __builtin_prefetch(aPtr + 64, 0, 1);
  __builtin_prefetch(bPtr[0] + 64, 0, 1);

  for (int k0 = 0; k0 < Kd; k0 += 64) {
    // issue loads for the odd half-step while buffer 0 computes
    load_frags(aPtr, bPtr, k0 + 32, a1f, b1f);
#pragma unroll
    for (int u = 0; u < NT; ++u)
      acc[u] = __builtin_amdgcn_wmma_f32_16x16x32_f16(
          false, a0f, false, b0f[u], (short)0, acc[u], false, false);

    // issue loads for the next even half-step (clamped; value unused if OOB)
    int k2 = (k0 + 64 < Kd) ? (k0 + 64) : k0;
    load_frags(aPtr, bPtr, k2, a0f, b0f);
#pragma unroll
    for (int u = 0; u < NT; ++u)
      acc[u] = __builtin_amdgcn_wmma_f32_16x16x32_f16(
          false, a1f, false, b1f[u], (short)0, acc[u], false, false);
  }

  // ---------------- epilogue
  int m0 = tm * 16 + hi * 8;                  // first of this lane's 8 C rows
  if (storeT) {
    // f16 transposed store: [N x M], lane's 8 rows are contiguous -> one b128
#pragma unroll
    for (int u = 0; u < NT; ++u) {
      int col = n0 + u * 16 + l16;
      float bv = bias ? bias[col] : 0.f;
      size_t ci = (size_t)batch * strideC + (size_t)col * ldcT + m0;
      if (m0 + 8 <= M) {
        v8h pk;
#pragma unroll
        for (int j = 0; j < 8; ++j) {
          float val = acc[u][j] * alpha + bv;
          if (doRelu) val = fmaxf(val, 0.f);
          pk[j] = (_Float16)val;
        }
        *(v8h*)(Ch + ci) = pk;
      } else {
#pragma unroll
        for (int j = 0; j < 8; ++j)
          if (m0 + j < M) {
            float val = acc[u][j] * alpha + bv;
            if (doRelu) val = fmaxf(val, 0.f);
            Ch[ci + j] = (_Float16)val;
          }
      }
    }
  } else {
#pragma unroll
    for (int u = 0; u < NT; ++u) {
      int col = n0 + u * 16 + l16;
      float bv = bias ? bias[col] : 0.f;
#pragma unroll
      for (int j = 0; j < 8; ++j) {
        int m = m0 + j;
        if (m < M) {
          float val = acc[u][j] * alpha + bv;
          if (doRelu) val = fmaxf(val, 0.f);
          size_t ci = (size_t)batch * strideC + (size_t)m * ldc + col;
          if (residual) val += residual[ci];
          if (Ch) Ch[ci] = (_Float16)val;
          else    C[ci]  = val;
        }
      }
    }
  }
}

// ---------------------------------------------------------------------------
// Elementwise / reduction kernels (f32 math, wave32 reductions)
// ---------------------------------------------------------------------------
__global__ void cast_f32_to_f16_kernel(const float* __restrict__ in,
                                       _Float16* __restrict__ out, int n) {
  int i = blockIdx.x * blockDim.x + threadIdx.x;
  if (i < n) out[i] = (_Float16)in[i];
}

// in: [R x Cd] row-major f32 -> out: [Cd x R] row-major f16 (transposed)
__global__ void castT_f32_to_f16_kernel(const float* __restrict__ in,
                                        _Float16* __restrict__ out, int R, int Cd) {
  int i = blockIdx.x * blockDim.x + threadIdx.x;
  if (i < R * Cd) {
    int r = i / Cd, c = i - r * Cd;
    out[(size_t)c * R + r] = (_Float16)in[i];
  }
}

__global__ void copy_f32_kernel(const float* __restrict__ in,
                                float* __restrict__ out, int n) {
  int i = blockIdx.x * blockDim.x + threadIdx.x;
  if (i < n) out[i] = in[i];
}

__global__ void init_slots_kernel(const float* __restrict__ z,
                                  const float* __restrict__ sigma,
                                  const float* __restrict__ w,
                                  float* __restrict__ slots, int n) {
  int i = blockIdx.x * blockDim.x + threadIdx.x;
  if (i < n) {
    int r = i % (Kc * Sc);
    float mu = w[r];
    slots[i] = mu + z[i] * sigma[0] * mu;
  }
}

// one wave per row; C multiple of 32; f16 output (GEMM A operand)
__global__ void ln_rows_f16_kernel(const float* __restrict__ x,
                                   const float* __restrict__ g,
                                   const float* __restrict__ b,
                                   _Float16* __restrict__ out, int C) {
  int row = blockIdx.x;
  int lane = threadIdx.x & 31;
  const float* xr = x + (size_t)row * C;
  float s = 0.f, s2 = 0.f;
  for (int i = lane; i < C; i += 32) { float v = xr[i]; s += v; s2 += v * v; }
  for (int o = 16; o > 0; o >>= 1) {
    s  += __shfl_xor(s,  o, 32);
    s2 += __shfl_xor(s2, o, 32);
  }
  float mu  = s / C;
  float var = s2 / C - mu * mu;
  float inv = rsqrtf(var + LN_EPS);
  _Float16* orow = out + (size_t)row * C;
  for (int i = lane; i < C; i += 32)
    orow[i] = (_Float16)((xr[i] - mu) * inv * g[i] + b[i]);
}

// softmax over the slot axis (Kc=11) for each (b, n) column
__global__ void softmax_slots_kernel(const float* __restrict__ logits,
                                     float* __restrict__ attn, int total) {
  int idx = blockIdx.x * blockDim.x + threadIdx.x;
  if (idx >= total) return;
  int b = idx / Nc, n = idx - b * Nc;
  const float* L = logits + (size_t)b * Kc * Nc + n;
  float e[Kc];
  float m = -3.4e38f;
#pragma unroll
  for (int k = 0; k < Kc; ++k) { float v = L[(size_t)k * Nc]; e[k] = v; m = fmaxf(m, v); }
  float s = 0.f;
#pragma unroll
  for (int k = 0; k < Kc; ++k) { float t = expf(e[k] - m); e[k] = t; s += t; }
  float inv = 1.f / s;
  float* Ao = attn + (size_t)b * Kc * Nc + n;
#pragma unroll
  for (int k = 0; k < Kc; ++k) Ao[(size_t)k * Nc] = e[k] * inv;
}

// one wave per (b,k) row: sum attn over N
__global__ void rowsum_kernel(const float* __restrict__ attn,
                              float* __restrict__ sums) {
  int row = blockIdx.x;
  int lane = threadIdx.x & 31;
  const float* a = attn + (size_t)row * Nc;
  float s = 0.f;
  for (int i = lane; i < Nc; i += 32) s += a[i];
  for (int o = 16; o > 0; o >>= 1) s += __shfl_xor(s, o, 32);
  if (lane == 0) sums[row] = s;
}

__global__ void attn_wm_kernel(const float* __restrict__ attn,
                               const float* __restrict__ sums,
                               _Float16* __restrict__ out, int total) {
  int i = blockIdx.x * blockDim.x + threadIdx.x;
  if (i < total) out[i] = (_Float16)(attn[i] / (sums[i / Nc] + SLOT_EPS));
}

// torch GRUCell combine: gates (r, z, n)
__global__ void gru_combine_kernel(const float* __restrict__ gi,
                                   const float* __restrict__ gh,
                                   const float* __restrict__ h,
                                   float* __restrict__ out, int rows) {
  int idx = blockIdx.x * blockDim.x + threadIdx.x;
  int total = rows * Sc;
  if (idx >= total) return;
  int i = idx / Sc, j = idx - i * Sc;
  const float* gir = gi + (size_t)i * 3 * Sc;
  const float* ghr = gh + (size_t)i * 3 * Sc;
  float r  = 1.f / (1.f + expf(-(gir[j]      + ghr[j])));
  float zg = 1.f / (1.f + expf(-(gir[Sc + j] + ghr[Sc + j])));
  float nn = tanhf(gir[2 * Sc + j] + r * ghr[2 * Sc + j]);
  out[idx] = (1.f - zg) * nn + zg * h[idx];
}

// ---------------------------------------------------------------------------
// Host side
// ---------------------------------------------------------------------------
static void launch_gemm(hipStream_t s,
                        const _Float16* A, int lda, long long strideA,
                        const _Float16* B, int ldb, long long strideB,
                        float* C, _Float16* Ch, int ldc, long long strideC,
                        int M, int N, int Kd, int batch, float alpha,
                        const float* bias, int relu, const float* residual,
                        int storeT = 0, int ldcT = 0) {
  int tilesM  = (M + 15) / 16;
  int groupsN = N / (NT * 16);          // N % 64 == 0 for every call site
  dim3 grid((unsigned)(tilesM * groupsN * batch));
  wmma_gemm_f16_kernel<<<grid, 32, 0, s>>>(A, lda, strideA, B, ldb, strideB,
                                           C, Ch, ldc, strideC,
                                           M, Kd, tilesM, groupsN,
                                           alpha, bias, relu, residual,
                                           storeT, ldcT);
}

extern "C" void kernel_launch(void* const* d_in, const int* in_sizes, int n_in,
                              void* d_out, int out_size, void* d_ws, size_t ws_size,
                              hipStream_t stream) {
  const float* features = (const float*)d_in[0];
  const float* sigma    = (const float*)d_in[1];
  const float* z        = (const float*)d_in[2];
  const float* slots_w  = (const float*)d_in[3];
  const float* gf = (const float*)d_in[4];
  const float* bf = (const float*)d_in[5];
  const float* gs = (const float*)d_in[6];
  const float* bs = (const float*)d_in[7];
  const float* gm = (const float*)d_in[8];
  const float* bm = (const float*)d_in[9];
  const float* Wk = (const float*)d_in[10];
  const float* Wv = (const float*)d_in[11];
  const float* Wq = (const float*)d_in[12];
  const float* W_ih = (const float*)d_in[13];
  const float* W_hh = (const float*)d_in[14];
  const float* b_ih = (const float*)d_in[15];
  const float* b_hh = (const float*)d_in[16];
  const float* W1 = (const float*)d_in[17];
  const float* b1 = (const float*)d_in[18];
  const float* W2 = (const float*)d_in[19];
  const float* b2 = (const float*)d_in[20];
  (void)in_sizes; (void)n_in; (void)out_size;

  float* outSlots = (float*)d_out;                      // [B,K,S]
  float* outAttn  = outSlots + (size_t)Bc * Kc * Sc;    // [B,K,N]

  char* base = (char*)d_ws;
  size_t off = 0;
  auto alloc = [&](size_t bytes) -> void* {
    off = (off + 255) & ~(size_t)255;
    void* p = base + off;
    off += bytes;
    return p;
  };
  (void)ws_size;

  // f16 activations / weights (T suffix = stored transposed, [N x Kd])
  _Float16* featsH = (_Float16*)alloc((size_t)Bc * Nc * Fc * 2);
  _Float16* kH     = (_Float16*)alloc((size_t)Bc * Nc * Sc * 2);  // [b][n][s]
  _Float16* vTH    = (_Float16*)alloc((size_t)Bc * Sc * Nc * 2);  // [b][s][n]
  _Float16* WkT    = (_Float16*)alloc((size_t)Sc * Fc * 2);
  _Float16* WvT    = (_Float16*)alloc((size_t)Sc * Fc * 2);
  _Float16* WqT    = (_Float16*)alloc((size_t)Sc * Sc * 2);
  _Float16* WihH   = (_Float16*)alloc((size_t)3 * Sc * Sc * 2);   // [3S x S] as-is
  _Float16* WhhH   = (_Float16*)alloc((size_t)3 * Sc * Sc * 2);
  _Float16* W1T    = (_Float16*)alloc((size_t)Mc * Sc * 2);
  _Float16* W2T    = (_Float16*)alloc((size_t)Sc * Mc * 2);
  _Float16* lnsH   = (_Float16*)alloc((size_t)Rc * Sc * 2);
  _Float16* qH     = (_Float16*)alloc((size_t)Rc * Sc * 2);
  _Float16* prevH  = (_Float16*)alloc((size_t)Rc * Sc * 2);
  _Float16* updH   = (_Float16*)alloc((size_t)Rc * Sc * 2);
  _Float16* wmH    = (_Float16*)alloc((size_t)Bc * Kc * Nc * 2);
  _Float16* lnmH   = (_Float16*)alloc((size_t)Rc * Sc * 2);
  _Float16* h1H    = (_Float16*)alloc((size_t)Rc * Mc * 2);
  // f32 scratch
  float* logitsF = (float*)alloc((size_t)Bc * Kc * Nc * 4);
  float* giF     = (float*)alloc((size_t)Rc * 3 * Sc * 4);
  float* ghF     = (float*)alloc((size_t)Rc * 3 * Sc * 4);
  float* sA      = (float*)alloc((size_t)Rc * Sc * 4);   // current slots
  float* sB      = (float*)alloc((size_t)Rc * Sc * 4);   // post-GRU slots
  float* sums    = (float*)alloc((size_t)Bc * Kc * 4);

  auto ew = [](int n) { return dim3((unsigned)((n + 255) / 256)); };

  // ---- weight conversion (f32 -> f16), transposing where the GEMM needs B^T
  castT_f32_to_f16_kernel<<<ew(Fc * Sc), 256, 0, stream>>>(Wk, WkT, Fc, Sc);
  castT_f32_to_f16_kernel<<<ew(Fc * Sc), 256, 0, stream>>>(Wv, WvT, Fc, Sc);
  castT_f32_to_f16_kernel<<<ew(Sc * Sc), 256, 0, stream>>>(Wq, WqT, Sc, Sc);
  cast_f32_to_f16_kernel<<<ew(3 * Sc * Sc), 256, 0, stream>>>(W_ih, WihH, 3 * Sc * Sc);
  cast_f32_to_f16_kernel<<<ew(3 * Sc * Sc), 256, 0, stream>>>(W_hh, WhhH, 3 * Sc * Sc);
  castT_f32_to_f16_kernel<<<ew(Sc * Mc), 256, 0, stream>>>(W1, W1T, Sc, Mc);
  castT_f32_to_f16_kernel<<<ew(Mc * Sc), 256, 0, stream>>>(W2, W2T, Mc, Sc);

  // ---- slots_init (stop-grad trick is a forward no-op)
  init_slots_kernel<<<ew(Rc * Sc), 256, 0, stream>>>(z, sigma, slots_w, sA, Rc * Sc);

  // ---- feats = LN(features); k = feats@Wk ; v^T = (feats@Wv)^T
  ln_rows_f16_kernel<<<Bc * Nc, 32, 0, stream>>>(features, gf, bf, featsH, Fc);
  launch_gemm(stream, featsH, Fc, 0, WkT, Fc, 0, nullptr, kH, Sc, 0,
              Bc * Nc, Sc, Fc, 1, 1.f, nullptr, 0, nullptr);
  launch_gemm(stream, featsH, Fc, (long long)Nc * Fc, WvT, Fc, 0,
              nullptr, vTH, 0, (long long)Nc * Sc,
              Nc, Sc, Fc, Bc, 1.f, nullptr, 0, nullptr,
              /*storeT=*/1, /*ldcT=*/Nc);

  const float scale = 1.f / sqrtf((float)Sc);

  for (int it = 0; it < 3; ++it) {
    // q = LN(slots)@Wq   (f16 out, GEMM A of logits)
    ln_rows_f16_kernel<<<Rc, 32, 0, stream>>>(sA, gs, bs, lnsH, Sc);
    launch_gemm(stream, lnsH, Sc, 0, WqT, Sc, 0, nullptr, qH, Sc, 0,
                Rc, Sc, Sc, 1, 1.f, nullptr, 0, nullptr);

    // logits[b] = scale * q[b] x k[b]^T   (batched, M=11 clamped to tile)
    launch_gemm(stream, qH, Sc, (long long)Kc * Sc,
                kH, Sc, (long long)Nc * Sc,
                logitsF, nullptr, Nc, (long long)Kc * Nc,
                Kc, Nc, Sc, Bc, scale, nullptr, 0, nullptr);

    // softmax over slots axis -> attn (written straight into d_out attn region)
    softmax_slots_kernel<<<ew(Bc * Nc), 256, 0, stream>>>(logitsF, outAttn, Bc * Nc);

    // weighted-mean renorm: attn_wm = attn / (rowsum + eps), f16
    rowsum_kernel<<<Bc * Kc, 32, 0, stream>>>(outAttn, sums);
    attn_wm_kernel<<<ew(Bc * Kc * Nc), 256, 0, stream>>>(outAttn, sums, wmH, Bc * Kc * Nc);

    // updates[b] = attn_wm[b] x v[b]  == wm x (v^T)^T  (f16 out)
    launch_gemm(stream, wmH, Nc, (long long)Kc * Nc,
                vTH, Nc, (long long)Nc * Sc,
                nullptr, updH, Sc, (long long)Kc * Sc,
                Kc, Sc, Nc, Bc, 1.f, nullptr, 0, nullptr);

    // GRU: gi = updates@W_ih^T + b_ih ; gh = slots_prev@W_hh^T + b_hh
    cast_f32_to_f16_kernel<<<ew(Rc * Sc), 256, 0, stream>>>(sA, prevH, Rc * Sc);
    launch_gemm(stream, updH, Sc, 0, WihH, Sc, 0, giF, nullptr, 3 * Sc, 0,
                Rc, 3 * Sc, Sc, 1, 1.f, b_ih, 0, nullptr);
    launch_gemm(stream, prevH, Sc, 0, WhhH, Sc, 0, ghF, nullptr, 3 * Sc, 0,
                Rc, 3 * Sc, Sc, 1, 1.f, b_hh, 0, nullptr);
    gru_combine_kernel<<<ew(Rc * Sc), 256, 0, stream>>>(giF, ghF, sA, sB, Rc);

    // MLP residual: slots = sB + relu(LN(sB)@W1 + b1)@W2 + b2
    ln_rows_f16_kernel<<<Rc, 32, 0, stream>>>(sB, gm, bm, lnmH, Sc);
    launch_gemm(stream, lnmH, Sc, 0, W1T, Sc, 0, nullptr, h1H, Mc, 0,
                Rc, Mc, Sc, 1, 1.f, b1, /*relu=*/1, nullptr);
    launch_gemm(stream, h1H, Mc, 0, W2T, Mc, 0, sA, nullptr, Sc, 0,
                Rc, Sc, Mc, 1, 1.f, b2, 0, /*residual=*/sB);
  }

  // final slots -> d_out slots region (attn already written by last softmax)
  copy_f32_kernel<<<ew(Rc * Sc), 256, 0, stream>>>(sA, outSlots, Rc * Sc);
}
